// GAT_large_no_edge_attr_6201932775765
// MI455X (gfx1250) — matrix operand order repaired and checked by reading, not compiled, not run on hardware
//
#include <hip/hip_runtime.h>

typedef __attribute__((ext_vector_type(16))) __bf16 v16bf;
typedef __attribute__((ext_vector_type(8)))  __bf16 v8bf;
typedef __attribute__((ext_vector_type(8)))  float  v8f;

#define NEG_SLOPE 0.2f

// ---------------------------------------------------------------------------
// One-time converts.
//   cvt_x  : X f32 [n,128]            -> Xb bf16 [n,128]
//   cvt_wt : W f32 [K=128][N=128]     -> Wt bf16 [N=128][K=128] (transposed)
// Transposed bf16 weights make each lane's WMMA B-fragment (two runs of 8
// K values at fixed output column) contiguous: 2x global_load_b128 per tile.
// ---------------------------------------------------------------------------
__global__ void cvt_x_bf16(const float* __restrict__ in, __bf16* __restrict__ out,
                           int total) {
    int idx = blockIdx.x * blockDim.x + threadIdx.x;
    if (idx < total) out[idx] = (__bf16)in[idx];
}

__global__ void cvt_w_transpose(const float* __restrict__ W, __bf16* __restrict__ Wt) {
    int idx = blockIdx.x * blockDim.x + threadIdx.x;   // 16384 threads
    int k = idx >> 7, ncol = idx & 127;                // coalesced read over n
    Wt[(size_t)ncol * 128 + k] = (__bf16)W[(size_t)k * 128 + ncol];
}

// ---------------------------------------------------------------------------
// WMMA GEMM: Y[n,128] = Xb[n,128] @ W[128,128]  (+bias, +relu optional)
// Xb bf16 row-major, Wt bf16 [N][K] (transposed). n % 16 == 0 (N=50000).
// Block = 256 threads (8 waves); wave w -> 16x16 tile (blockIdx.x*16, w*16).
// Per K-step: 4x b128 loads + 1x v_wmma_f32_16x16x32_bf16. No guards.
// A 16x32 bf16 lane layout: lanes 0-15 M=lane,K-half 0; lanes 16-31 K-half 8.
// B 32x16 bf16 lane layout: N=lane%16, K 0-7,16-23 (lanes<16) / 8-15,24-31 (lanes>=16).
// C/D: VGPR r -> M=r (+8 for upper lanes), N=lane%16.
// ---------------------------------------------------------------------------
__global__ void gat_wmma_gemm(const __bf16* __restrict__ Xb,
                              const __bf16* __restrict__ Wt,
                              const float* __restrict__ bias,
                              float* __restrict__ Y,
                              int n, int relu_flag) {
    const int lane = threadIdx.x & 31;
    const int wave = threadIdx.x >> 5;           // 0..7 -> N-tile
    const int m0   = blockIdx.x * 16;
    const int col0 = wave * 16;
    const int half = lane >> 4;
    const int mrow = m0 + (lane & 15);
    const int bcol = col0 + (lane & 15);

    const __bf16* ar = Xb + (size_t)mrow * 128 + half * 8;    // A K-offset
    const __bf16* br = Wt + (size_t)bcol * 128 + half * 8;    // B K-offset

    v8f acc = {};
    #pragma unroll
    for (int k0 = 0; k0 < 128; k0 += 32) {
        v8bf alo = *(const v8bf*)(ar + k0);        // K = k0+aK   .. +7
        v8bf ahi = *(const v8bf*)(ar + k0 + 16);   // K = k0+16+aK.. +7
        v8bf blo = *(const v8bf*)(br + k0);        // K = k0+bK   .. +7
        v8bf bhi = *(const v8bf*)(br + k0 + 16);   // K = k0+16+bK.. +7
        v16bf a = __builtin_shufflevector(alo, ahi, 0,1,2,3,4,5,6,7,
                                          8,9,10,11,12,13,14,15);
        v16bf b = __builtin_shufflevector(blo, bhi, 0,1,2,3,4,5,6,7,
                                          8,9,10,11,12,13,14,15);
        acc = __builtin_amdgcn_wmma_f32_16x16x32_bf16(false, a, false, b,
                                                      (short)0, acc, false, false);
    }

    float* yr = Y + (size_t)(m0 + half * 8) * 128 + bcol;
    #pragma unroll
    for (int r = 0; r < 8; ++r) {
        float v = acc[r];
        if (bias) v += bias[bcol];
        if (relu_flag) v = fmaxf(v, 0.0f);
        yr[(size_t)r * 128] = v;
    }
}

// ---------------------------------------------------------------------------
// Attention logits: al_s[n,h] = <h[n,h,:], a_src[h,:]>, same for al_d.
// ---------------------------------------------------------------------------
__global__ void gat_attn_logits(const float* __restrict__ hbuf,
                                const float* __restrict__ a_src,
                                const float* __restrict__ a_dst,
                                float* __restrict__ al_s,
                                float* __restrict__ al_d,
                                int n, int H, int C) {
    int idx = blockIdx.x * blockDim.x + threadIdx.x;
    if (idx >= n * H) return;
    int node = idx / H, h = idx % H;
    const float* hp = hbuf + (size_t)node * 128 + h * C;
    const float* as = a_src + h * C;
    const float* ad = a_dst + h * C;
    float ss = 0.0f, sd = 0.0f;
    for (int c = 0; c < C; ++c) { float v = hp[c]; ss += v * as[c]; sd += v * ad[c]; }
    al_s[idx] = ss;
    al_d[idx] = sd;
}

// ---------------------------------------------------------------------------
// Per-layer init: agg = 0, emax = -inf, denom = 0.
// ---------------------------------------------------------------------------
__global__ void gat_init(float* __restrict__ agg,
                         float* __restrict__ emax,
                         float* __restrict__ denom,
                         int n, int H) {
    int idx = blockIdx.x * blockDim.x + threadIdx.x;
    if (idx >= n * 128) return;
    agg[idx] = 0.0f;
    int node = idx >> 7, j = idx & 127;
    if (j < H) {
        emax[node * H + j]  = -__builtin_inff();
        denom[node * H + j] = 0.0f;
    }
}

__device__ __forceinline__ float gat_leaky(float e) {
    return e > 0.0f ? e : NEG_SLOPE * e;
}

// atomic float max via sign-aware int atomics (native VMEM atomics, no CAS loop)
__device__ __forceinline__ void atomicMaxF(float* addr, float v) {
    if (v >= 0.0f) atomicMax((int*)addr, __float_as_int(v));
    else           atomicMin((unsigned int*)addr, __float_as_uint(v));
}

// ---------------------------------------------------------------------------
// Edge pass 1: segment max of leakyrelu(al_s[src]+al_d[dst]) over dst.
// Edges [0,E) from edge_index; [E, E+n) are self loops.
// ---------------------------------------------------------------------------
__global__ void gat_edge_max(const int* __restrict__ src,
                             const int* __restrict__ dst,
                             const float* __restrict__ al_s,
                             const float* __restrict__ al_d,
                             float* __restrict__ emax,
                             int E, int n, int H) {
    int idx = blockIdx.x * blockDim.x + threadIdx.x;
    int ET = E + n;
    if (idx >= ET * H) return;
    int eid = idx / H, h = idx % H;
    int s = (eid < E) ? src[eid] : (eid - E);
    int d = (eid < E) ? dst[eid] : (eid - E);
    float e = gat_leaky(al_s[s * H + h] + al_d[d * H + h]);
    atomicMaxF(&emax[d * H + h], e);
}

// ---------------------------------------------------------------------------
// Edge pass 2: denom[dst,h] += exp(e - emax[dst,h])
// ---------------------------------------------------------------------------
__global__ void gat_edge_sum(const int* __restrict__ src,
                             const int* __restrict__ dst,
                             const float* __restrict__ al_s,
                             const float* __restrict__ al_d,
                             const float* __restrict__ emax,
                             float* __restrict__ denom,
                             int E, int n, int H) {
    int idx = blockIdx.x * blockDim.x + threadIdx.x;
    int ET = E + n;
    if (idx >= ET * H) return;
    int eid = idx / H, h = idx % H;
    int s = (eid < E) ? src[eid] : (eid - E);
    int d = (eid < E) ? dst[eid] : (eid - E);
    float e = gat_leaky(al_s[s * H + h] + al_d[d * H + h]);
    atomicAdd(&denom[d * H + h], __expf(e - emax[d * H + h]));
}

// ---------------------------------------------------------------------------
// Edge pass 3: agg[dst, :] += h[src, :] * alpha. One thread = one 16-channel
// chunk of one edge (8 chunks cover 128 channels; head = chunk*16/C).
// float4 gathers + per-float global atomic adds (bandwidth bound).
// ---------------------------------------------------------------------------
__global__ void gat_edge_agg(const int* __restrict__ src,
                             const int* __restrict__ dst,
                             const float* __restrict__ al_s,
                             const float* __restrict__ al_d,
                             const float* __restrict__ emax,
                             const float* __restrict__ denom,
                             const float* __restrict__ hbuf,
                             float* __restrict__ agg,
                             int E, int n, int H, int C) {
    int idx = blockIdx.x * blockDim.x + threadIdx.x;
    int ET = E + n;
    if (idx >= ET * 8) return;
    int eid = idx >> 3, q = idx & 7;
    int s = (eid < E) ? src[eid] : (eid - E);
    int d = (eid < E) ? dst[eid] : (eid - E);
    int h = (q * 16) / C;
    float e = gat_leaky(al_s[s * H + h] + al_d[d * H + h]);
    float alpha = __expf(e - emax[d * H + h]) / denom[d * H + h];
    const float4* hp = (const float4*)(hbuf + (size_t)s * 128 + q * 16);
    float* ap = agg + (size_t)d * 128 + q * 16;
    #pragma unroll
    for (int v = 0; v < 4; ++v) {
        float4 hv = hp[v];
        atomicAdd(ap + v * 4 + 0, hv.x * alpha);
        atomicAdd(ap + v * 4 + 1, hv.y * alpha);
        atomicAdd(ap + v * 4 + 2, hv.z * alpha);
        atomicAdd(ap + v * 4 + 3, hv.w * alpha);
    }
}

// ---------------------------------------------------------------------------
// Finalize: x_next(bf16) = relu(agg + b). Next consumer is only the next
// GEMM's A operand, so write bf16 directly (halves the traffic too).
// ---------------------------------------------------------------------------
__global__ void gat_bias_relu_bf16(const float* __restrict__ agg,
                                   const float* __restrict__ b,
                                   __bf16* __restrict__ out,
                                   int total) {
    int idx = blockIdx.x * blockDim.x + threadIdx.x;
    if (idx >= total) return;
    out[idx] = (__bf16)fmaxf(agg[idx] + b[idx & 127], 0.0f);
}

// ---------------------------------------------------------------------------
extern "C" void kernel_launch(void* const* d_in, const int* in_sizes, int n_in,
                              void* d_out, int out_size, void* d_ws, size_t ws_size,
                              hipStream_t stream) {
    const float* x  = (const float*)d_in[0];
    const int*   ei = (const int*)d_in[1];
    const int n = in_sizes[0] / 128;
    const int E = in_sizes[1] / 2;
    const int* src = ei;
    const int* dst = ei + E;

    // layer params: W, a_src, a_dst, b at indices 3..22; W_lin=23, b_lin=24
    const float* W[6];  const float* AS[5]; const float* AD[5]; const float* B[5];
    for (int l = 0; l < 5; ++l) {
        W[l]  = (const float*)d_in[3 + 4 * l + 0];
        AS[l] = (const float*)d_in[3 + 4 * l + 1];
        AD[l] = (const float*)d_in[3 + 4 * l + 2];
        B[l]  = (const float*)d_in[3 + 4 * l + 3];
    }
    W[5] = (const float*)d_in[23];              // W_lin
    const float* b_lin = (const float*)d_in[24];
    float* out = (float*)d_out;

    // workspace carve: hbuf(f32) | agg(f32) | xb(bf16) | al_s | al_d | emax |
    //                  denom | Wt[6](bf16, transposed)
    char* p = (char*)d_ws;
    float*  hbuf = (float*)p;  p += (size_t)n * 128 * sizeof(float);
    float*  agg  = (float*)p;  p += (size_t)n * 128 * sizeof(float);
    __bf16* xb   = (__bf16*)p; p += (size_t)n * 128 * sizeof(__bf16);
    float*  al_s = (float*)p;  p += (size_t)n * 8 * sizeof(float);
    float*  al_d = (float*)p;  p += (size_t)n * 8 * sizeof(float);
    float*  emax = (float*)p;  p += (size_t)n * 8 * sizeof(float);
    float*  denom= (float*)p;  p += (size_t)n * 8 * sizeof(float);
    __bf16* Wt[6];
    for (int l = 0; l < 6; ++l) { Wt[l] = (__bf16*)p; p += 128 * 128 * sizeof(__bf16); }

    const int TB = 256;
    const int mtiles = n / 16;                   // n % 16 == 0 (N = 50000)
    const int ET = E + n;
    const int nodeTot = n * 128;

    // one-time operand preparation (bf16 activations, transposed bf16 weights)
    cvt_x_bf16<<<(nodeTot + TB - 1) / TB, TB, 0, stream>>>(x, xb, nodeTot);
    for (int l = 0; l < 6; ++l)
        cvt_w_transpose<<<(16384 + TB - 1) / TB, TB, 0, stream>>>(W[l], Wt[l]);

    auto layer = [&](int l, int H, int C) {
        gat_wmma_gemm<<<mtiles, TB, 0, stream>>>(xb, Wt[l], nullptr, hbuf, n, 0);
        int t1 = n * H;
        gat_attn_logits<<<(t1 + TB - 1) / TB, TB, 0, stream>>>(hbuf, AS[l], AD[l],
                                                               al_s, al_d, n, H, C);
        gat_init<<<(nodeTot + TB - 1) / TB, TB, 0, stream>>>(agg, emax, denom, n, H);
        int t2 = ET * H;
        gat_edge_max<<<(t2 + TB - 1) / TB, TB, 0, stream>>>(src, dst, al_s, al_d,
                                                            emax, E, n, H);
        gat_edge_sum<<<(t2 + TB - 1) / TB, TB, 0, stream>>>(src, dst, al_s, al_d,
                                                            emax, denom, E, n, H);
        int t3 = ET * 8;
        gat_edge_agg<<<(t3 + TB - 1) / TB, TB, 0, stream>>>(src, dst, al_s, al_d,
                                                            emax, denom, hbuf, agg,
                                                            E, n, H, C);
        gat_bias_relu_bf16<<<(nodeTot + TB - 1) / TB, TB, 0, stream>>>(agg, B[l],
                                                                       xb, nodeTot);
    };

    layer(0, 8, 16);    // conv1
    layer(1, 8, 16);    // conv2
    layer(2, 8, 16);    // conv3
    layer(3, 8, 16);    // conv4
    layer(4, 1, 128);   // conv5 (single head)

    // final linear + bias + relu fused into WMMA epilogue (f32 out -> d_out)
    gat_wmma_gemm<<<mtiles, TB, 0, stream>>>(xb, Wt[5], b_lin, out, n, 1);
}
